// LightweightSelfAttention_9603546874129
// MI455X (gfx1250) — compile-verified, matrix operand order checked by reading, not run on hardware
//
#include <hip/hip_runtime.h>
#include <hip/hip_bf16.h>

#define CDIM   512
#define NHEADS 8
#define NTOK   4096   // 64*64
#define DH     64

typedef __attribute__((ext_vector_type(16))) _Float16 v16h;
typedef __attribute__((ext_vector_type(8)))  _Float16 v8h;
typedef __attribute__((ext_vector_type(8)))  float    v8f;

// Load one 16-lane-striped WMMA fragment (16 f16 per lane) from a row-major
// f16 buffer. Caller passes p = base + row*ld + hi*8; layout per CDNA5 ISA:
// lanes 0-15 hold K = {0..7, 16..23}, lanes 16-31 hold K = {8..15, 24..31}.
__device__ __forceinline__ v16h ldfrag(const _Float16* p) {
  v8h lo = *(const v8h*)(p);
  v8h hh = *(const v8h*)(p + 16);
  v16h r;
#pragma unroll
  for (int i = 0; i < 8; ++i) { r[i] = lo[i]; r[i + 8] = hh[i]; }
  return r;
}

__device__ __forceinline__ v8f wmma_f16(v16h a, v16h b, v8f c) {
  // D = A(16x32 f16) * B(32x16 f16) + C(16x16 f32)
  return __builtin_amdgcn_wmma_f32_16x16x32_f16(false, a, false, b, (short)0, c,
                                                false, false);
}

// ---------------------------------------------------------------- K0: weights
__global__ void __launch_bounds__(256)
convert_w_kernel(const float* __restrict__ qkvw, const float* __restrict__ projw,
                 _Float16* __restrict__ wq, _Float16* __restrict__ wp) {
  int i = blockIdx.x * 256 + threadIdx.x;
  if (i < 3 * CDIM * CDIM) {
    float v = qkvw[i];
    if (i < CDIM * CDIM) v *= 0.125f;  // fold dh^-0.5 into W_q
    wq[i] = (_Float16)v;
  }
  if (i < CDIM * CDIM) wp[i] = (_Float16)projw[i];
}

// ------------------------------------------------------- K1: depthwise 3x3 conv
// x (B,C,64,64) fp32 -> yT (B,N,C) f16  (transposed so GEMM B-operand is
// contiguous in the channel/K dimension)
__global__ void __launch_bounds__(256)
dwconv_kernel(const float* __restrict__ x, const float* __restrict__ dw,
              _Float16* __restrict__ yT) {
  int idx = blockIdx.x * 256 + threadIdx.x;       // B*C*N = 4194304 threads
  int n = idx & (NTOK - 1);
  int c = (idx >> 12) & (CDIM - 1);
  int b = idx >> 21;
  int hh = n >> 6, ww = n & 63;
  const float* xb = x + ((size_t)(b * CDIM + c) << 12);
  const float* wf = dw + c * 9;
  float s = 0.f;
#pragma unroll
  for (int dy = -1; dy <= 1; ++dy) {
    int yy = hh + dy;
    if ((unsigned)yy >= 64u) continue;
#pragma unroll
    for (int dx = -1; dx <= 1; ++dx) {
      int xx = ww + dx;
      if ((unsigned)xx >= 64u) continue;
      s += xb[yy * 64 + xx] * wf[(dy + 1) * 3 + (dx + 1)];
    }
  }
  yT[(size_t)(b * NTOK + n) * CDIM + c] = (_Float16)s;
}

// ------------------------------------------------------------ K2: qkv 1x1 GEMM
// out(3C x N) = Wqkv(3C x C) * yT^T; one wave per 32x32 register-blocked tile
// (2x2 fragments, 4 accumulators -> 1 WMMA per fragment load).
// q,k stored row-major (B,H,N,dh); v stored transposed (B,H,dh,N).
__global__ void __launch_bounds__(256)
qkv_kernel(const _Float16* __restrict__ wq, const _Float16* __restrict__ yT,
           _Float16* __restrict__ qb, _Float16* __restrict__ kb,
           _Float16* __restrict__ vtb) {
  const int wid  = blockIdx.x * 8 + (threadIdx.x >> 5);
  const int lane = threadIdx.x & 31;
  const int li = lane & 15, hi = lane >> 4;
  const int b   = wid / (48 * 128);
  const int rem = wid % (48 * 128);
  const int o0  = (rem / 128) * 32;   // over 3C = 1536
  const int n0  = (rem % 128) * 32;   // over N  = 4096
  const _Float16* Y = yT + (size_t)b * NTOK * CDIM;

  v8f acc[2][2] = {{{}, {}}, {{}, {}}};
  for (int kk = 0; kk < CDIM; kk += 32) {
    v16h a0 = ldfrag(wq + (size_t)(o0 + li) * CDIM + kk + hi * 8);
    v16h a1 = ldfrag(wq + (size_t)(o0 + 16 + li) * CDIM + kk + hi * 8);
    v16h b0 = ldfrag(Y + (size_t)(n0 + li) * CDIM + kk + hi * 8);
    v16h b1 = ldfrag(Y + (size_t)(n0 + 16 + li) * CDIM + kk + hi * 8);
    acc[0][0] = wmma_f16(a0, b0, acc[0][0]);
    acc[0][1] = wmma_f16(a0, b1, acc[0][1]);
    acc[1][0] = wmma_f16(a1, b0, acc[1][0]);
    acc[1][1] = wmma_f16(a1, b1, acc[1][1]);
  }

  const int which = o0 >> 9;          // 0=q 1=k 2=v (tile never crosses: 512%32==0)
  const int h = (o0 & 511) >> 6;      // tile never crosses a head (64%32==0)
#pragma unroll
  for (int io = 0; io < 2; ++io) {
    const int d0 = ((o0 + 16 * io) & 63) + 8 * hi;  // C-layout: row m = r + 8*hi
#pragma unroll
    for (int in = 0; in < 2; ++in) {
      const int n = n0 + 16 * in + li;
      if (which < 2) {
        v8h pk;
#pragma unroll
        for (int r = 0; r < 8; ++r) pk[r] = (_Float16)acc[io][in][r];
        _Float16* dst = (which == 0 ? qb : kb) +
                        (size_t)((b * NHEADS + h) * NTOK + n) * DH + d0;
        *(v8h*)dst = pk;
      } else {
#pragma unroll
        for (int r = 0; r < 8; ++r)
          vtb[(size_t)((b * NHEADS + h) * DH + d0 + r) * NTOK + n] =
              (_Float16)acc[io][in][r];
      }
    }
  }
}

// ---------------------------------------------------- K3: flash attention (f16)
// One wave = 16 q-rows; online softmax; P transposed via per-wave LDS slice.
__global__ void __launch_bounds__(256)
attn_kernel(const _Float16* __restrict__ qb, const _Float16* __restrict__ kb,
            const _Float16* __restrict__ vtb, _Float16* __restrict__ otb) {
  __shared__ _Float16 plds[8 * 16 * 32];  // 8 waves x (16x32) P tile
  const int w    = threadIdx.x >> 5;
  const int lane = threadIdx.x & 31;
  const int li = lane & 15, hi = lane >> 4;

  const int t  = blockIdx.x;       // 512 blocks = B * H * (N/128)
  const int nb = t & 31;
  const int h  = (t >> 5) & 7;
  const int b  = t >> 8;
  const int m0 = nb * 128 + w * 16;

  const _Float16* Q  = qb  + (size_t)((b * NHEADS + h) * NTOK) * DH;
  const _Float16* K  = kb  + (size_t)((b * NHEADS + h) * NTOK) * DH;
  const _Float16* Vt = vtb + (size_t)((b * NHEADS + h) * DH) * NTOK;
  _Float16* my = plds + w * 512;

  const v16h qa0 = ldfrag(Q + (m0 + li) * DH + hi * 8);        // dh 0..31
  const v16h qa1 = ldfrag(Q + (m0 + li) * DH + 32 + hi * 8);   // dh 32..63

  v8f o0 = {}, o1 = {}, o2 = {}, o3 = {};
  float mr[8], lr[8];
#pragma unroll
  for (int r = 0; r < 8; ++r) { mr[r] = -3.0e38f; lr[r] = 0.0f; }

  for (int j = 0; j < NTOK; j += 32) {
    // Prefetch next key block's K rows (one 128B row per lane) and V lines into
    // L2/WGP$ -> global_prefetch_b8. Stays inside the workspace allocation.
    __builtin_prefetch(K + (size_t)(j + 32 + lane) * DH, 0, 1);
    __builtin_prefetch(Vt + (size_t)lane * NTOK + j + 32, 0, 1);
    __builtin_prefetch(Vt + (size_t)(32 + lane) * NTOK + j + 32, 0, 1);

    v8f z = {};
    v16h k0 = ldfrag(K + (j + li) * DH + hi * 8);
    v16h k1 = ldfrag(K + (j + li) * DH + 32 + hi * 8);
    v8f s0 = wmma_f16(qa0, k0, z);
    s0 = wmma_f16(qa1, k1, s0);                         // keys j..j+15
    v16h k2 = ldfrag(K + (j + 16 + li) * DH + hi * 8);
    v16h k3 = ldfrag(K + (j + 16 + li) * DH + 32 + hi * 8);
    v8f s1 = wmma_f16(qa0, k2, z);
    s1 = wmma_f16(qa1, k3, s1);                         // keys j+16..j+31

#pragma unroll
    for (int r = 0; r < 8; ++r) {       // row m = r + 8*hi lives in 16 lanes
      float mx = fmaxf(s0[r], s1[r]);
#pragma unroll
      for (int msk = 1; msk < 16; msk <<= 1) mx = fmaxf(mx, __shfl_xor(mx, msk));
      float mnew  = fmaxf(mr[r], mx);
      float alpha = __expf(mr[r] - mnew);
      float p0 = __expf(s0[r] - mnew);
      float p1 = __expf(s1[r] - mnew);
      float rs = p0 + p1;
#pragma unroll
      for (int msk = 1; msk < 16; msk <<= 1) rs += __shfl_xor(rs, msk);
      lr[r] = lr[r] * alpha + rs;
      mr[r] = mnew;
      my[(r + 8 * hi) * 32 + li]      = (_Float16)p0;   // C-layout -> row-major
      my[(r + 8 * hi) * 32 + 16 + li] = (_Float16)p1;
      o0[r] *= alpha; o1[r] *= alpha; o2[r] *= alpha; o3[r] *= alpha;
    }
    asm volatile("s_wait_dscnt 0" ::: "memory");  // P writes visible (same wave)
    v16h pa = ldfrag(my + li * 32 + hi * 8);      // re-read as A-fragment
    v16h v0 = ldfrag(Vt + (0 * 16 + li) * NTOK + j + hi * 8);
    o0 = wmma_f16(pa, v0, o0);
    v16h v1 = ldfrag(Vt + (1 * 16 + li) * NTOK + j + hi * 8);
    o1 = wmma_f16(pa, v1, o1);
    v16h v2 = ldfrag(Vt + (2 * 16 + li) * NTOK + j + hi * 8);
    o2 = wmma_f16(pa, v2, o2);
    v16h v3 = ldfrag(Vt + (3 * 16 + li) * NTOK + j + hi * 8);
    o3 = wmma_f16(pa, v3, o3);
  }

  // store O^T as Ot (B, N, C) f16, c = h*64 + d  (matches torch transpose+reshape)
#pragma unroll
  for (int r = 0; r < 8; ++r) {
    float inv = 1.0f / lr[r];
    int row = m0 + r + 8 * hi;
    _Float16* op = otb + (size_t)(b * NTOK + row) * CDIM + h * DH + li;
    op[0]  = (_Float16)(o0[r] * inv);
    op[16] = (_Float16)(o1[r] * inv);
    op[32] = (_Float16)(o2[r] * inv);
    op[48] = (_Float16)(o3[r] * inv);
  }
}

// ----------------------------------------------------------- K4: proj GEMM+bias
// 32x32 register-blocked tiles, fp32 output + bias.
__global__ void __launch_bounds__(256)
proj_kernel(const _Float16* __restrict__ wp, const _Float16* __restrict__ Ot,
            const float* __restrict__ pb, float* __restrict__ out) {
  const int wid  = blockIdx.x * 8 + (threadIdx.x >> 5);
  const int lane = threadIdx.x & 31;
  const int li = lane & 15, hi = lane >> 4;
  const int b   = wid / (16 * 128);
  const int rem = wid % (16 * 128);
  const int o0  = (rem / 128) * 32;
  const int n0  = (rem % 128) * 32;
  const _Float16* O = Ot + (size_t)b * NTOK * CDIM;

  v8f acc[2][2] = {{{}, {}}, {{}, {}}};
  for (int kk = 0; kk < CDIM; kk += 32) {
    v16h a0 = ldfrag(wp + (size_t)(o0 + li) * CDIM + kk + hi * 8);
    v16h a1 = ldfrag(wp + (size_t)(o0 + 16 + li) * CDIM + kk + hi * 8);
    v16h b0 = ldfrag(O + (size_t)(n0 + li) * CDIM + kk + hi * 8);
    v16h b1 = ldfrag(O + (size_t)(n0 + 16 + li) * CDIM + kk + hi * 8);
    acc[0][0] = wmma_f16(a0, b0, acc[0][0]);
    acc[0][1] = wmma_f16(a0, b1, acc[0][1]);
    acc[1][0] = wmma_f16(a1, b0, acc[1][0]);
    acc[1][1] = wmma_f16(a1, b1, acc[1][1]);
  }

#pragma unroll
  for (int io = 0; io < 2; ++io) {
#pragma unroll
    for (int in = 0; in < 2; ++in) {
      const int n = n0 + 16 * in + li;
#pragma unroll
      for (int r = 0; r < 8; ++r) {
        int oc = o0 + 16 * io + r + 8 * hi;
        out[(size_t)(b * CDIM + oc) * NTOK + n] = acc[io][in][r] + pb[oc];
      }
    }
  }
}

// ------------------------------------------------------------------- launcher
extern "C" void kernel_launch(void* const* d_in, const int* in_sizes, int n_in,
                              void* d_out, int out_size, void* d_ws, size_t ws_size,
                              hipStream_t stream) {
  const float* x     = (const float*)d_in[0];
  const float* dww   = (const float*)d_in[1];
  const float* qkvw  = (const float*)d_in[2];
  const float* projw = (const float*)d_in[3];
  const float* projb = (const float*)d_in[4];
  float* out = (float*)d_out;

  _Float16* ws = (_Float16*)d_ws;
  size_t off = 0;
  _Float16* wq = ws + off; off += (size_t)3 * CDIM * CDIM;          // 786432
  _Float16* wp = ws + off; off += (size_t)CDIM * CDIM;              // 262144
  _Float16* yT = ws + off; off += (size_t)2 * NTOK * CDIM;          // 4.19M
  _Float16* qb = ws + off; off += (size_t)2 * NHEADS * NTOK * DH;   // 4.19M
  _Float16* kb = ws + off; off += (size_t)2 * NHEADS * NTOK * DH;
  _Float16* vt = ws + off; off += (size_t)2 * NHEADS * NTOK * DH;
  _Float16* ot = ws + off; off += (size_t)2 * NTOK * CDIM;
  if (ws_size < off * sizeof(_Float16)) return;  // ~42 MB required

  convert_w_kernel<<<3072, 256, 0, stream>>>(qkvw, projw, wq, wp);
  dwconv_kernel<<<16384, 256, 0, stream>>>(x, dww, yT);
  qkv_kernel<<<1536, 256, 0, stream>>>(wq, yT, qb, kb, vt);
  attn_kernel<<<512, 256, 0, stream>>>(qb, kb, vt, ot);
  proj_kernel<<<512, 256, 0, stream>>>(wp, ot, projb, out);
}
